// GritNetWithCustomBLSTM_40785009443238
// MI455X (gfx1250) — compile-verified, matrix operand order checked by reading, not run on hardware
//
#include <hip/hip_runtime.h>

// ---------------------------------------------------------------------------
// GritNet BLSTM for MI455X (gfx1250, wave32, WMMA bf16 -> f32 accumulate)
//
// Pipeline:
//   1) convert emb / W_ih halves / W_hh to bf16 in workspace
//   2) table GEMM (WMMA): P[t][1113,512] = emb @ W_half.T (+ baked biases)
//      (emb[event_x] @ W.T == (emb @ W.T)[event_x]  -> 9.3 GFLOP instead of
//       137 GFLOP and no 268 MB materialized x; tables are L2-resident)
//   3) LSTM recurrence: 1 workgroup per direction, everything LDS-resident,
//      per-step [32,512] = h[32,128] x Whh^T[128,512] via v_wmma bf16,
//      table rows prefetched (global_prefetch_b8) behind the step GEMM,
//      gate nonlinearities + running max-over-time in LDS
//   4) output head: sigmoid(maxpool @ W_out.T + b_out)
// ---------------------------------------------------------------------------

typedef __attribute__((ext_vector_type(16))) __bf16 v16bf;
typedef __attribute__((ext_vector_type(8)))  float  v8f;

#define EVENT_DIM 1113
#define EMBED_DIM 2048
#define HIDDEN    128
#define BB        32
#define TT        512
#define PROWS     1120   // 1113 padded to 70 tiles of 16

// workspace layout (bytes)
#define OFF_EMBBF  0u
#define SZ_EMBBF   (1113u*2048u*2u)                 // 4,558,848
#define OFF_WIHBF  (OFF_EMBBF + SZ_EMBBF)
#define SZ_WIHBF   (4u*512u*2048u*2u)               // 8,388,608
#define OFF_WHHBF  (OFF_WIHBF + SZ_WIHBF)
#define SZ_WHHBF   (2u*512u*128u*2u)                // 262,144
#define OFF_CB     (OFF_WHHBF + SZ_WHHBF)
#define SZ_CB      (4u*512u*4u)                     // 8,192
#define OFF_P      (OFF_CB + SZ_CB)
#define SZ_P       (4u*1120u*512u*4u)               // 9,175,040
#define OFF_GMAX   (OFF_P + SZ_P)
#define SZ_GMAX    (2u*32u*128u*4u)                 // 32,768

// ---- WMMA helpers ---------------------------------------------------------

__device__ __forceinline__ v8f wmma_bf16(v16bf a, v16bf b, v8f c) {
    // D = A(16x32 bf16) * B(32x16 bf16) + C(16x16 f32)
    return __builtin_amdgcn_wmma_f32_16x16x32_bf16(
        /*neg_a=*/false, a, /*neg_b=*/false, b,
        /*c_mod=*/(short)0, c, /*reuse_a=*/false, /*reuse_b=*/false);
}

// Load one 16-lane operand slice in the ISA 16-bit A/B layout from a
// row-major [rows, K] bf16 matrix. Caller passes a per-lane pointer at
// (row = base + (lane&15), col = kbase + kh) with kh = ((lane>>4)<<3).
// Elements 0..7  <- K = kh+kbase+0..7
// Elements 8..15 <- K = kh+kbase+16..23
__device__ __forceinline__ v16bf ldpair(const __bf16* q) {
    union U { uint4 u[2]; v16bf v; } x;
    x.u[0] = *(const uint4*)(q);
    x.u[1] = *(const uint4*)(q + 16);
    return x.v;
}

// ---- prep kernels ---------------------------------------------------------

__global__ void cvt_f32_bf16(const float* __restrict__ src, __bf16* __restrict__ dst, int n) {
    int i = blockIdx.x * 256 + threadIdx.x;
    if (i < n) dst[i] = (__bf16)src[i];
}

// dst[4][512][2048]: t0 = W_ih_f[:, :2048], t1 = W_ih_f[:, 2048:],
//                    t2 = W_ih_b[:, :2048], t3 = W_ih_b[:, 2048:]
__global__ void build_wih(const float* __restrict__ Wf, const float* __restrict__ Wb,
                          __bf16* __restrict__ dst) {
    int i = blockIdx.x * 256 + threadIdx.x;
    if (i >= 4 * 512 * 2048) return;
    int k = i & 2047;
    int n = (i >> 11) & 511;
    int t = i >> 20;
    const float* W = (t < 2) ? Wf : Wb;
    int half = t & 1;
    dst[i] = (__bf16)W[n * 4096 + half * 2048 + k];
}

// combined per-gate bias baked into the event_x tables (t0, t2)
__global__ void build_bias(const float* __restrict__ bihf, const float* __restrict__ bhhf,
                           const float* __restrict__ bihb, const float* __restrict__ bhhb,
                           float* __restrict__ cb) {
    int i = blockIdx.x * 256 + threadIdx.x;
    if (i >= 4 * 512) return;
    int n = i & 511, t = i >> 9;
    float v = 0.f;
    if (t == 0) v = bihf[n] + bhhf[n];
    else if (t == 2) v = bihb[n] + bhhb[n];
    cb[i] = v;
}

// ---- table GEMM: P[t] = emb(bf16) @ Wt^T, f32 accum, bias epilogue --------
// grid = (16 npair, 70 mtile, 4 table), block = 32 (one wave -> 2 N-tiles)

__global__ __launch_bounds__(32) void table_gemm(
    const __bf16* __restrict__ embbf, const __bf16* __restrict__ Wbf,
    const float* __restrict__ cb, float* __restrict__ P) {
    const int lane  = threadIdx.x & 31;
    const int npair = blockIdx.x;
    const int mtile = blockIdx.y;
    const int tbl   = blockIdx.z;

    const __bf16* W    = Wbf + (size_t)tbl * (512u * 2048u);
    float*        Pout = P   + (size_t)tbl * (PROWS * 512u);

    int rA = mtile * 16 + (lane & 15);
    if (rA > EVENT_DIM - 1) rA = EVENT_DIM - 1;   // clamp pad rows (never read)
    const int kh = (lane >> 4) << 3;

    const __bf16* arow  = embbf + (size_t)rA * 2048 + kh;
    const int n0 = npair * 32, n1 = n0 + 16;
    const __bf16* brow0 = W + (size_t)(n0 + (lane & 15)) * 2048 + kh;
    const __bf16* brow1 = W + (size_t)(n1 + (lane & 15)) * 2048 + kh;

    v8f acc0 = {}, acc1 = {};
    for (int k = 0; k < 2048; k += 32) {
        v16bf a  = ldpair(arow + k);
        v16bf b0 = ldpair(brow0 + k);
        v16bf b1 = ldpair(brow1 + k);
        acc0 = wmma_bf16(a, b0, acc0);
        acc1 = wmma_bf16(a, b1, acc1);
    }

    const float bias0 = cb[tbl * 512 + n0 + (lane & 15)];
    const float bias1 = cb[tbl * 512 + n1 + (lane & 15)];
    const int   m0    = mtile * 16 + ((lane >> 4) << 3);
    const int   nc0   = n0 + (lane & 15), nc1 = n1 + (lane & 15);
#pragma unroll
    for (int j = 0; j < 8; ++j) {
        Pout[(size_t)(m0 + j) * 512 + nc0] = acc0[j] + bias0;
        Pout[(size_t)(m0 + j) * 512 + nc1] = acc1[j] + bias1;
    }
}

// ---- LSTM recurrence: grid=2 (direction), block=512 (16 waves) ------------

#define HSTR 136   // padded bf16 row stride (kills 16-row ds_read_b128 bank conflicts)

__global__ __launch_bounds__(512) void lstm_kernel(
    const int* __restrict__ event_x, const int* __restrict__ event_time_x,
    const float* __restrict__ P, const __bf16* __restrict__ whhbf,
    float* __restrict__ gmax) {
    __shared__ __align__(16) __bf16 Wl[512 * HSTR];   // W_hh bf16  (~136 KB)
    __shared__ __align__(16) __bf16 hl[32 * HSTR];    // h state bf16
    __shared__ float cl[32 * 128];                    // c state
    __shared__ float hm[32 * 128];                    // running max over time
    __shared__ float gl[32 * 512];                    // gate preacts (WMMA out)
    __shared__ int   exs[32], ets[32];

    const int tid  = threadIdx.x;
    const int dir  = blockIdx.x;          // 0 = fwd, 1 = bwd
    const int lane = tid & 31;
    const int wave = tid >> 5;            // 0..15

    // init: stage W_hh into LDS (padded), zero states
    const __bf16* Wg = whhbf + (size_t)dir * 512 * 128;
    for (int i = tid; i < 512 * 128; i += 512) {
        int n = i >> 7, k = i & 127;
        Wl[n * HSTR + k] = Wg[i];
    }
    for (int i = tid; i < 32 * HSTR; i += 512) hl[i] = (__bf16)0.f;
    for (int i = tid; i < 32 * 128; i += 512) { cl[i] = 0.f; hm[i] = -1e30f; }
    __syncthreads();

    const float* P0 = P + (size_t)(dir * 2) * (PROWS * 512u);  // event_x table (+bias)
    const float* P1 = P0 + (PROWS * 512u);                     // event_time_x table

    // per-wave WMMA tile assignment: ntiles {2w, 2w+1} x mtiles {0,1}
    const int n0 = wave * 32, n1 = n0 + 16;
    const int kh = (lane >> 4) << 3;
    const __bf16* b0p = &Wl[(size_t)(n0 + (lane & 15)) * HSTR + kh];
    const __bf16* b1p = &Wl[(size_t)(n1 + (lane & 15)) * HSTR + kh];
    const __bf16* a0p = &hl[(size_t)(lane & 15) * HSTR + kh];
    const __bf16* a1p = &hl[(size_t)(16 + (lane & 15)) * HSTR + kh];
    const int ms   = (lane >> 4) << 3;    // D-tile row base within tile
    const int ncol = lane & 15;

    // prefetch segment assignment: 16 threads per batch row, 128B segments
    const int pb  = tid >> 4;             // 0..31 batch row
    const int pseg = (tid & 15) * 32;     // 16 x 32 floats = full 2KB row

    for (int t = 0; t < TT; ++t) {
        const int te = dir ? (TT - 1 - t) : t;   // max over time is order-invariant
        if (tid < 32) {
            exs[tid] = event_x[tid * TT + te];
            ets[tid] = event_time_x[tid * TT + te];
        }

        // cooperatively prefetch this step's 64 table rows (2KB each) into the
        // WGP cache so the gate phase's reads hit near-cache; the step GEMM
        // below provides the latency cover. global_prefetch_b8: no counters.
        {
            const int e0 = event_x[pb * TT + te];
            const int e1 = event_time_x[pb * TT + te];
            __builtin_prefetch(P0 + (size_t)e0 * 512 + pseg, 0, 3);
            __builtin_prefetch(P1 + (size_t)e1 * 512 + pseg, 0, 3);
        }

        // step GEMM: g = h @ Whh^T  ([32,512], K=128), bf16 in / f32 accum
        v8f c00 = {}, c01 = {}, c10 = {}, c11 = {};
#pragma unroll
        for (int k = 0; k < 128; k += 32) {
            v16bf a0 = ldpair(a0p + k);
            v16bf a1 = ldpair(a1p + k);
            v16bf b0 = ldpair(b0p + k);
            v16bf b1 = ldpair(b1p + k);
            c00 = wmma_bf16(a0, b0, c00);
            c01 = wmma_bf16(a0, b1, c01);
            c10 = wmma_bf16(a1, b0, c10);
            c11 = wmma_bf16(a1, b1, c11);
        }
#pragma unroll
        for (int j = 0; j < 8; ++j) {
            gl[(ms + j) * 512 + n0 + ncol]      = c00[j];
            gl[(ms + j) * 512 + n1 + ncol]      = c01[j];
            gl[(16 + ms + j) * 512 + n0 + ncol] = c10[j];
            gl[(16 + ms + j) * 512 + n1 + ncol] = c11[j];
        }
        __syncthreads();

        // gates + state update + running max: 4096 (b,h) pairs
        for (int p = tid; p < 32 * 128; p += 512) {
            const int b = p >> 7, hh = p & 127;
            const float* r0 = P0 + (size_t)exs[b] * 512;   // prefetched rows
            const float* r1 = P1 + (size_t)ets[b] * 512;
            const float gi = gl[b * 512 + hh]        + r0[hh]        + r1[hh];
            const float gf = gl[b * 512 + 128 + hh]  + r0[128 + hh]  + r1[128 + hh];
            const float gc = gl[b * 512 + 256 + hh]  + r0[256 + hh]  + r1[256 + hh];
            const float go = gl[b * 512 + 384 + hh]  + r0[384 + hh]  + r1[384 + hh];
            const float si = 1.f / (1.f + __expf(-gi));
            const float sf = 1.f / (1.f + __expf(-gf));
            const float so = 1.f / (1.f + __expf(-go));
            const float c  = sf * cl[p] + si * tanhf(gc);
            const float h  = so * tanhf(c);
            cl[p] = c;
            hl[b * HSTR + hh] = (__bf16)h;
            hm[p] = fmaxf(hm[p], h);
        }
        __syncthreads();
    }

    for (int p = tid; p < 32 * 128; p += 512)
        gmax[dir * 4096 + p] = hm[p];
}

// ---- output head ----------------------------------------------------------

__global__ void out_kernel(const float* __restrict__ gmax, const float* __restrict__ Wout,
                           const float* __restrict__ bout, float* __restrict__ out) {
    int b = threadIdx.x;
    if (b >= 32) return;
    float s = bout[0];
    for (int j = 0; j < 128; ++j) {
        s += gmax[b * 128 + j]        * Wout[j];         // forward half
        s += gmax[4096 + b * 128 + j] * Wout[128 + j];   // backward half
    }
    out[b] = 1.f / (1.f + __expf(-s));
}

// ---- launcher -------------------------------------------------------------

extern "C" void kernel_launch(void* const* d_in, const int* in_sizes, int n_in,
                              void* d_out, int out_size, void* d_ws, size_t ws_size,
                              hipStream_t stream) {
    (void)in_sizes; (void)n_in; (void)out_size; (void)ws_size;
    const int*   ex   = (const int*)d_in[0];
    const int*   et   = (const int*)d_in[1];
    /* d_in[2] = lens: unused by the reference computation */
    const float* emb  = (const float*)d_in[3];
    const float* Wihf = (const float*)d_in[4];
    const float* bihf = (const float*)d_in[5];
    const float* Whhf = (const float*)d_in[6];
    const float* bhhf = (const float*)d_in[7];
    const float* Wihb = (const float*)d_in[8];
    const float* bihb = (const float*)d_in[9];
    const float* Whhb = (const float*)d_in[10];
    const float* bhhb = (const float*)d_in[11];
    const float* Wout = (const float*)d_in[12];
    const float* bout = (const float*)d_in[13];
    float*       out  = (float*)d_out;

    char* ws = (char*)d_ws;
    __bf16* embbf = (__bf16*)(ws + OFF_EMBBF);
    __bf16* wihbf = (__bf16*)(ws + OFF_WIHBF);
    __bf16* whhbf = (__bf16*)(ws + OFF_WHHBF);
    float*  cb    = (float*)(ws + OFF_CB);
    float*  Ptab  = (float*)(ws + OFF_P);
    float*  gmx   = (float*)(ws + OFF_GMAX);

    const int nEmb = EVENT_DIM * EMBED_DIM;         // 2,279,424
    const int nWih = 4 * 512 * 2048;                // 4,194,304
    const int nWhh = 512 * 128;                     // 65,536

    cvt_f32_bf16<<<(nEmb + 255) / 256, 256, 0, stream>>>(emb, embbf, nEmb);
    build_wih  <<<(nWih + 255) / 256, 256, 0, stream>>>(Wihf, Wihb, wihbf);
    cvt_f32_bf16<<<(nWhh + 255) / 256, 256, 0, stream>>>(Whhf, whhbf, nWhh);
    cvt_f32_bf16<<<(nWhh + 255) / 256, 256, 0, stream>>>(Whhb, whhbf + nWhh, nWhh);
    build_bias <<<8, 256, 0, stream>>>(bihf, bhhf, bihb, bhhb, cb);

    table_gemm <<<dim3(16, 70, 4), 32, 0, stream>>>(embbf, wihbf, cb, Ptab);
    lstm_kernel<<<2, 512, 0, stream>>>(ex, et, Ptab, whhbf, gmx);
    out_kernel <<<1, 32, 0, stream>>>(gmx, Wout, bout, out);
}
